// MultiModalModel_21071109554673
// MI455X (gfx1250) — matrix operand order repaired and checked by reading, not compiled, not run on hardware
//
#include <hip/hip_runtime.h>
#include <hip/hip_bf16.h>

#define Bsz  16
#define NA_  512
#define NV_  256
#define D_   512
#define KC   32     // K-chunk per LDS stage (one bf16 WMMA K-depth)
#define KP   40     // padded LDS row stride (elements): 80B rows, 16B aligned, conflict-free

typedef __attribute__((ext_vector_type(16))) __bf16 v16bf;
typedef __attribute__((ext_vector_type(8)))  __bf16 v8bf;
typedef __attribute__((ext_vector_type(8)))  float  v8f;

union BF8 { v8bf v; __bf16 e[8]; };

// A-matrix 16x32 bf16 fragment (ISA 7.12.2): lanes 0-15 get K pairs {0..7,16..23},
// lanes 16-31 get {8..15,24..31}; source is row-major [row][k] in LDS.
__device__ __forceinline__ v16bf load_fragA(const __bf16* base, int row, int lane) {
  const int kOff = (lane >> 4) * 8;
  const __bf16* p = base + row * KP + kOff;
  v8bf a = *(const v8bf*)p;
  v8bf b = *(const v8bf*)(p + 16);
  return __builtin_shufflevector(a, b, 0,1,2,3,4,5,6,7,8,9,10,11,12,13,14,15);
}

// B-matrix 32x16 bf16 fragment: lane = N column; lanes 0-15 hold K={0..7,16..23},
// lanes 16-31 hold K={8..15,24..31}. Source row-major V[m][k] == (B^T) rows.
__device__ __forceinline__ v16bf load_fragB(const __bf16* base, int row, int lane) {
  const int kOff = (lane >> 4) * 8;
  const __bf16* p = base + row * KP + kOff;
  v8bf a = *(const v8bf*)p;
  v8bf b = *(const v8bf*)(p + 16);
  return __builtin_shufflevector(a, b, 0,1,2,3,4,5,6,7,8,9,10,11,12,13,14,15);
}

__global__ __launch_bounds__(256)
void colbert_sim_kernel(const float* __restrict__ audio,
                        const float* __restrict__ visual,
                        const float* __restrict__ tptr,
                        float* __restrict__ ws)
{
  __shared__ __align__(16) __bf16 ldsAhi[128 * KP];
  __shared__ __align__(16) __bf16 ldsAlo[128 * KP];
  __shared__ __align__(16) __bf16 ldsVhi[256 * KP];
  __shared__ __align__(16) __bf16 ldsVlo[256 * KP];
  __shared__ float redClip[16];
  __shared__ float redNN[8];

  const int strip = blockIdx.x;            // 0..3  -> 128-row n strip
  const int c     = blockIdx.y;            // visual batch
  const int b     = blockIdx.z;            // audio batch
  const int tid   = threadIdx.x;
  const int lane  = tid & 31;
  const int wave  = tid >> 5;
  const int nBase = strip * 128;

  const float invTemp = 1.0f / tptr[0];

  v8f acc[16];
  const v8f vzero = {0.f,0.f,0.f,0.f,0.f,0.f,0.f,0.f};
#pragma unroll
  for (int mt = 0; mt < 16; ++mt) acc[mt] = vzero;

  // LDS fill assignments
  const int rowAf = tid >> 1;                       // 0..127
  const int cgA   = (tid & 1) * 16;                 // 16 floats each
  const float* gA_base = audio  + ((size_t)(b * NA_) + nBase + rowAf) * D_ + cgA;
  const float* gV_base = visual + ((size_t)(c * NV_) + tid) * D_;     // 32 floats each

  const int rowA = wave * 16 + (lane & 15);         // this wave's A-frag row

  for (int kc = 0; kc < D_; kc += KC) {
    __syncthreads();
    { // stage A strip (128 x 32) as bf16 hi/lo split
      const float* g = gA_base + kc;
      BF8 h[2], l[2];
#pragma unroll
      for (int i = 0; i < 16; ++i) {
        float x  = g[i];
        __bf16 hb = (__bf16)x;
        float r  = x - (float)hb;      // exact residual
        h[i >> 3].e[i & 7] = hb;
        l[i >> 3].e[i & 7] = (__bf16)r;
      }
      __bf16* dh = &ldsAhi[rowAf * KP + cgA];
      __bf16* dl = &ldsAlo[rowAf * KP + cgA];
      *(v8bf*)dh = h[0].v;  *(v8bf*)(dh + 8) = h[1].v;
      *(v8bf*)dl = l[0].v;  *(v8bf*)(dl + 8) = l[1].v;
    }
    { // stage V panel (256 x 32) as bf16 hi/lo split
      const float* g = gV_base + kc;
      BF8 h[4], l[4];
#pragma unroll
      for (int i = 0; i < 32; ++i) {
        float x  = g[i];
        __bf16 hb = (__bf16)x;
        float r  = x - (float)hb;
        h[i >> 3].e[i & 7] = hb;
        l[i >> 3].e[i & 7] = (__bf16)r;
      }
      __bf16* dh = &ldsVhi[tid * KP];
      __bf16* dl = &ldsVlo[tid * KP];
#pragma unroll
      for (int q = 0; q < 4; ++q) {
        *(v8bf*)(dh + 8 * q) = h[q].v;
        *(v8bf*)(dl + 8 * q) = l[q].v;
      }
    }
    __syncthreads();

    v16bf ahi = load_fragA(ldsAhi, rowA, lane);
    v16bf alo = load_fragA(ldsAlo, rowA, lane);
#pragma unroll
    for (int mt = 0; mt < 16; ++mt) {
      const int rowV = mt * 16 + (lane & 15);
      v16bf bhi = load_fragB(ldsVhi, rowV, lane);
      v16bf blo = load_fragB(ldsVlo, rowV, lane);
      // bf16x3: hi*hi + hi*lo + lo*hi, f32 accumulate (near-fp32 precision)
      acc[mt] = __builtin_amdgcn_wmma_f32_16x16x32_bf16(false, ahi, false, bhi, (short)0, acc[mt], false, false);
      acc[mt] = __builtin_amdgcn_wmma_f32_16x16x32_bf16(false, ahi, false, blo, (short)0, acc[mt], false, false);
      acc[mt] = __builtin_amdgcn_wmma_f32_16x16x32_bf16(false, alo, false, bhi, (short)0, acc[mt], false, false);
    }
  }

  // Fused epilogue: row-max over m (wave-local), clip(-20,0)^4 sum
  float rmax[8];
#pragma unroll
  for (int j = 0; j < 8; ++j) rmax[j] = -3.402823466e+38f;
  float acc4 = 0.0f;
#pragma unroll
  for (int mt = 0; mt < 16; ++mt) {
#pragma unroll
    for (int j = 0; j < 8; ++j) {
      float s = acc[mt][j] * invTemp;
      rmax[j] = fmaxf(rmax[j], s);
      float t = fminf(s, 0.0f);
      t = fmaxf(t, -20.0f);
      float t2 = t * t;
      acc4 += t2 * t2;
    }
  }
  // reduce max across the 16-lane N group (C layout: lane = N, VGPR j = M row)
#pragma unroll
  for (int j = 0; j < 8; ++j) {
#pragma unroll
    for (int m = 1; m < 16; m <<= 1)
      rmax[j] = fmaxf(rmax[j], __shfl_xor(rmax[j], m, 32));
  }
  float rowsum = 0.0f;
#pragma unroll
  for (int j = 0; j < 8; ++j) rowsum += rmax[j];   // lane0: rows 0..7, lane16: rows 8..15
#pragma unroll
  for (int m = 1; m < 32; m <<= 1) acc4 += __shfl_xor(acc4, m, 32);

  if ((lane & 15) == 0) redClip[wave * 2 + (lane >> 4)] = rowsum;
  if (lane == 0)        redNN[wave] = acc4;
  __syncthreads();
  if (tid == 0) {       // deterministic per-block partials, no atomics
    float cs = 0.f;
#pragma unroll
    for (int i = 0; i < 16; ++i) cs += redClip[i];
    float nn = 0.f;
#pragma unroll
    for (int i = 0; i < 8; ++i) nn += redNN[i];
    const int idx = (b * 16 + c) * 4 + strip;
    ws[idx]        = cs;   // sum over this strip's 128 row-maxes (already /temp)
    ws[1024 + idx] = nn;   // clip^4 partial
  }
}

__global__ void colbert_finalize(const float* __restrict__ ws,
                                 const float* __restrict__ tptr,
                                 float* __restrict__ out)
{
  if (threadIdx.x != 0 || blockIdx.x != 0) return;
  float clip[16][16];
  for (int i = 0; i < 16; ++i)
    for (int j = 0; j < 16; ++j) {
      const float* p = ws + (size_t)(i * 16 + j) * 4;
      clip[i][j] = (p[0] + p[1] + p[2] + p[3]) * (1.0f / 512.0f);
    }
  float closs = 0.f;
  for (int i = 0; i < 16; ++i) {
    float mr = -3.402823466e+38f, mc = -3.402823466e+38f;
    for (int j = 0; j < 16; ++j) {
      mr = fmaxf(mr, clip[i][j]);
      mc = fmaxf(mc, clip[j][i]);
    }
    float sr = 0.f, sc = 0.f;
    for (int j = 0; j < 16; ++j) {
      sr += expf(clip[i][j] - mr);
      sc += expf(clip[j][i] - mc);
    }
    closs += (mr + logf(sr) - clip[i][i]) + (mc + logf(sc) - clip[i][i]);
  }
  closs *= (1.0f / 16.0f) * 0.5f;

  double nn = 0.0;
  for (int k = 0; k < 1024; ++k) nn += (double)ws[1024 + k];
  float l_nonneg = (float)(nn / 33554432.0);     // 16*16*512*256

  float temp = tptr[0];
  float lt = logf(temp);
  float tl = fmaxf(-lt, 0.f);                  tl = tl * tl; tl = tl * tl;
  float th = fmaxf(lt - 1.3862943611198906f, 0.f); th = th * th; th = th * th;
  out[0] = closs + 8.0f * (tl + th) + 0.15f * l_nonneg;
}

extern "C" void kernel_launch(void* const* d_in, const int* in_sizes, int n_in,
                              void* d_out, int out_size, void* d_ws, size_t ws_size,
                              hipStream_t stream) {
  (void)in_sizes; (void)n_in; (void)out_size; (void)ws_size;
  const float* audio  = (const float*)d_in[0];
  const float* visual = (const float*)d_in[1];
  const float* tptr   = (const float*)d_in[2];
  float* out = (float*)d_out;
  float* ws  = (float*)d_ws;                   // needs 2048 floats (8 KB)

  dim3 grid(4, 16, 16);                        // (n-strip, c, b) = 1024 workgroups
  colbert_sim_kernel<<<grid, 256, 0, stream>>>(audio, visual, tptr, ws);
  colbert_finalize<<<1, 32, 0, stream>>>(ws, tptr, out);
}
